// HOI_Mamba_block_72619307041198
// MI455X (gfx1250) — matrix operand order repaired and checked by reading, not compiled, not run on hardware
//
#include <hip/hip_runtime.h>
#include <hip/hip_bf16.h>
#include <math.h>

// ---------- problem constants ----------
#define BB 8
#define OO 12
#define TT 64
#define DMQ 256
#define NLQ 2
#define DIQ 512
#define DSQ 16
#define DCONVQ 4
#define DTRQ 16
#define LQ (OO * TT)          // 768
#define MPC (BB * LQ)         // 6144 rows per combo
#define NCOMBO 4

// ---------- WMMA types ----------
typedef __attribute__((ext_vector_type(16))) __bf16 v16bf;
typedef __attribute__((ext_vector_type(8)))  float  v8f;

union BF16x16 { v16bf v; uint4 q[2]; unsigned short u[16]; };

__device__ __forceinline__ unsigned short f2bf(float f) {
    unsigned int u = __float_as_uint(f);
    unsigned int r = u + 0x7FFFu + ((u >> 16) & 1u);   // round-to-nearest-even
    return (unsigned short)(r >> 16);
}
__device__ __forceinline__ unsigned int pack2(float a, float b) {
    return (unsigned int)f2bf(a) | ((unsigned int)f2bf(b) << 16);
}

__device__ __forceinline__ float siluf(float x) { return x / (1.f + __expf(-x)); }
__device__ __forceinline__ float softplusf(float x) { return x > 20.f ? x : log1pf(__expf(x)); }

// ---------- elementwise: x + pe ----------
__global__ void add_pe_kernel(const float* __restrict__ x, const float* __restrict__ pe,
                              float* __restrict__ out, int n) {
    int i = blockIdx.x * blockDim.x + threadIdx.x;
    if (i >= n) return;
    int d = i % DMQ;
    int t = (i / DMQ) % TT;
    out[i] = x[i] + pe[t * DMQ + d];
}

// ---------- gather per-combo sequences (handles blk reshape + dir reversal) ----------
__global__ void gather_seq_kernel(const float* __restrict__ x, float* __restrict__ seq) {
    long i = (long)blockIdx.x * blockDim.x + threadIdx.x;
    long total = (long)NCOMBO * BB * LQ * DMQ;
    if (i >= total) return;
    int d = (int)(i % DMQ);
    int n = (int)((i / DMQ) % LQ);
    int b = (int)((i / ((long)DMQ * LQ)) % BB);
    int c = (int)(i / ((long)DMQ * LQ * BB));
    int dir = c & 1, blk = c >> 1;
    int m = dir ? (LQ - 1 - n) : n;
    int o, t;
    if (blk == 0) { o = m / TT; t = m % TT; }
    else          { t = m / OO; o = m % OO; }
    seq[i] = x[(((long)b * OO + o) * TT + t) * DMQ + d];
}

// ---------- generic bf16 WMMA GEMM: C = act(A @ W + bias), combo-batched on grid.z ----------
// A tile staged in LDS in *fragment order* so each lane's 16-bf16 A fragment is a
// contiguous 16B-aligned 32B run (W: two 16B runs) -> two ds_load_b128 per fragment.
__global__ void __launch_bounds__(128)
gemm_wmma_kernel(const float* __restrict__ A, long aStride, int lda,
                 const float* __restrict__ W, long wStride,
                 const float* __restrict__ bias, long biasStride,
                 float* __restrict__ C, long cStride, int ldc,
                 int N, int K, int act) {
    int combo = blockIdx.z;
    A += (long)combo * aStride;
    W += (long)combo * wStride;
    if (bias) bias += (long)combo * biasStride;
    C += (long)combo * cStride;

    // AFrag[row][h][16]: for half h the fragment order is k in [8h,8h+8) ++ [16+8h,16+8h+8)
    __shared__ __align__(16) unsigned short AFrag[64][2][16];
    // WFrag[n][k]: lane n, half h reads k in [8h,8h+8) ++ [16+8h,16+8h+8)
    __shared__ __align__(16) unsigned short WFrag[64][32];

    const int tid  = threadIdx.x;
    const int row0 = blockIdx.y * 64;
    const int col0 = blockIdx.x * 64;

    const int wave = tid >> 5;
    const int lane = tid & 31;
    const int half = lane >> 4;
    const int lr   = lane & 15;
    const int wm   = (wave >> 1) * 32;
    const int wn   = (wave & 1) * 32;

    v8f acc[2][2];
#pragma unroll
    for (int mi = 0; mi < 2; ++mi)
#pragma unroll
        for (int ni = 0; ni < 2; ++ni)
            acc[mi][ni] = v8f{};

    for (int kt = 0; kt < K; kt += 32) {
        // ---- stage A tile 64x32 -> fragment order (K always multiple of 16) ----
        {
            int r  = tid >> 1;
            int kc = (tid & 1) * 16;           // this thread covers k = kt+kc .. kt+kc+15
            float4 f0, f1, f2, f3;
            if (kt + kc < K) {                 // uniform 16-element group guard
                const float4* ap = reinterpret_cast<const float4*>(
                    A + (long)(row0 + r) * lda + kt + kc);
                f0 = ap[0]; f1 = ap[1]; f2 = ap[2]; f3 = ap[3];
            } else {
                f0 = f1 = f2 = f3 = float4{0.f, 0.f, 0.f, 0.f};
            }
            // elements j=0..7  -> AFrag[r][0][kc/2 .. kc/2+7]
            // elements j=8..15 -> AFrag[r][1][kc/2 .. kc/2+7]
            uint4 q0 = make_uint4(pack2(f0.x, f0.y), pack2(f0.z, f0.w),
                                  pack2(f1.x, f1.y), pack2(f1.z, f1.w));
            uint4 q1 = make_uint4(pack2(f2.x, f2.y), pack2(f2.z, f2.w),
                                  pack2(f3.x, f3.y), pack2(f3.z, f3.w));
            *reinterpret_cast<uint4*>(&AFrag[r][0][kc >> 1]) = q0;
            *reinterpret_cast<uint4*>(&AFrag[r][1][kc >> 1]) = q1;
        }
        // ---- stage W tile 32x64 transposed (branch-free clamp+mask loads) ----
        {
            int kk = tid >> 6;                 // 0..1  (k block of 16)
            int n  = tid & 63;
            int gn = col0 + n;
            int nsafe = (gn < N) ? gn : (N - 1);
            float nm = (gn < N) ? 1.f : 0.f;
            unsigned int q[8];
#pragma unroll
            for (int j = 0; j < 8; ++j) {
                int k0 = kt + kk * 16 + 2 * j;
                int k1 = k0 + 1;
                int k0s = (k0 < K) ? k0 : (K - 1);
                int k1s = (k1 < K) ? k1 : (K - 1);
                float v0 = W[(long)k0s * N + nsafe] * ((k0 < K) ? nm : 0.f);
                float v1 = W[(long)k1s * N + nsafe] * ((k1 < K) ? nm : 0.f);
                q[j] = pack2(v0, v1);
            }
            *reinterpret_cast<uint4*>(&WFrag[n][kk * 16])     = make_uint4(q[0], q[1], q[2], q[3]);
            *reinterpret_cast<uint4*>(&WFrag[n][kk * 16 + 8]) = make_uint4(q[4], q[5], q[6], q[7]);
        }
        __syncthreads();

        // ---- fragment loads: 2x ds_load_b128 each ----
        BF16x16 af[2], bfr[2];
#pragma unroll
        for (int mi = 0; mi < 2; ++mi) {
            int m = wm + mi * 16 + lr;
            const uint4* p = reinterpret_cast<const uint4*>(&AFrag[m][half][0]);
            af[mi].q[0] = p[0];
            af[mi].q[1] = p[1];
        }
#pragma unroll
        for (int ni = 0; ni < 2; ++ni) {
            int n = wn + ni * 16 + lr;
            const uint4* p = reinterpret_cast<const uint4*>(&WFrag[n][half * 8]);
            bfr[ni].q[0] = p[0];
            bfr[ni].q[1] = p[2];
        }
#pragma unroll
        for (int mi = 0; mi < 2; ++mi)
#pragma unroll
            for (int ni = 0; ni < 2; ++ni)
                acc[mi][ni] = __builtin_amdgcn_wmma_f32_16x16x32_bf16(
                    false, af[mi].v, false, bfr[ni].v, (short)0, acc[mi][ni], false, false);
        __syncthreads();
    }

    // epilogue: C layout — VGPR r: lanes0-15 M=r, lanes16-31 M=8+r
#pragma unroll
    for (int mi = 0; mi < 2; ++mi)
#pragma unroll
        for (int ni = 0; ni < 2; ++ni) {
            int colBase = col0 + wn + ni * 16 + lr;
            if (colBase < N) {
                float bv = bias ? bias[colBase] : 0.f;
#pragma unroll
                for (int r = 0; r < 8; ++r) {
                    int row = row0 + wm + mi * 16 + half * 8 + r;
                    float v = acc[mi][ni][r] + bv;
                    if (act == 1) v = softplusf(v);
                    C[(long)row * ldc + colBase] = v;
                }
            }
        }
}

// ---------- causal depthwise conv (K=4) + bias + SiLU ----------
__global__ void conv_silu_kernel(const float* __restrict__ xz, const float* __restrict__ cw,
                                 const float* __restrict__ cb, float* __restrict__ xc) {
    long i = (long)blockIdx.x * blockDim.x + threadIdx.x;
    long total = (long)NCOMBO * BB * LQ * DIQ;
    if (i >= total) return;
    int d = (int)(i % DIQ);
    int l = (int)((i / DIQ) % LQ);
    long row = i / DIQ;          // (c*8+b)*768 + l
    long seqBase = row - l;      // start of this (c,b) sequence
    int c = (int)(row / ((long)BB * LQ));
    const float* w = cw + ((long)c * DIQ + d) * DCONVQ;
    float acc = cb[(long)c * DIQ + d];
#pragma unroll
    for (int j = 0; j < DCONVQ; ++j) {
        int ls = l - (DCONVQ - 1) + j;
        if (ls >= 0) acc += w[j] * xz[(seqBase + ls) * (2 * DIQ) + d];
    }
    xc[i] = siluf(acc);
}

// ---------- selective scan (sequential over L, per-(c,b,d) state in registers) ----------
__global__ void __launch_bounds__(256)
scan_kernel(const float* __restrict__ delta, const float* __restrict__ xc,
            const float* __restrict__ dbl, const float* __restrict__ A_log,
            const float* __restrict__ Dp, float* __restrict__ y) {
    int blk = blockIdx.x;                 // 64 blocks: (c,b,dhalf)
    int dh = blk & 1;
    int cb = blk >> 1;
    int b = cb & 7;
    int c = cb >> 3;
    int d = dh * 256 + threadIdx.x;

    const float* al = A_log + ((long)c * DIQ + d) * DSQ;
    float A[DSQ];
#pragma unroll
    for (int s = 0; s < DSQ; ++s) A[s] = -__expf(al[s]);
    float Dd = Dp[(long)c * DIQ + d];

    float h[DSQ];
#pragma unroll
    for (int s = 0; s < DSQ; ++s) h[s] = 0.f;

    __shared__ float sBC[32];
    long base = (long)(c * BB + b) * LQ;

    for (int l = 0; l < LQ; ++l) {
        if (threadIdx.x < 32)
            sBC[threadIdx.x] = dbl[(base + l) * (DTRQ + 2 * DSQ) + DTRQ + threadIdx.x];
        __syncthreads();
        float dl = delta[(base + l) * DIQ + d];
        float xv = xc[(base + l) * DIQ + d];
        float dx = dl * xv;
        float yv = 0.f;
#pragma unroll
        for (int s = 0; s < DSQ; ++s) {
            h[s] = __expf(dl * A[s]) * h[s] + dx * sBC[s];
            yv += h[s] * sBC[DSQ + s];
        }
        y[(base + l) * DIQ + d] = yv + xv * Dd;
        __syncthreads();
    }
}

// ---------- y *= silu(z) ----------
__global__ void gate_kernel(const float* __restrict__ xz, float* __restrict__ y) {
    long i = (long)blockIdx.x * blockDim.x + threadIdx.x;
    long total = (long)NCOMBO * BB * LQ * DIQ;
    if (i >= total) return;
    int d = (int)(i % DIQ);
    long row = i / DIQ;
    float z = xz[row * (2 * DIQ) + DIQ + d];
    y[i] = y[i] * siluf(z);
}

// ---------- combine fwd+bwd, double layernorm, average, residual (in-place on x) ----------
__global__ void __launch_bounds__(256)
combine_ln_kernel(const float* __restrict__ dirout, const float* __restrict__ lnw,
                  const float* __restrict__ lnb, float* __restrict__ x) {
    int g = blockIdx.x;                       // B*O*T blocks
    int t = g % TT;
    int o = (g / TT) % OO;
    int b = g / (TT * OO);
    int d = threadIdx.x;

    long n0 = (long)o * TT + t;
    long n1 = (long)t * OO + o;
    long r0 = ((long)(0 * BB + b) * LQ + n0)            * DMQ;
    long r1 = ((long)(1 * BB + b) * LQ + (LQ - 1 - n0)) * DMQ;
    long r2 = ((long)(2 * BB + b) * LQ + n1)            * DMQ;
    long r3 = ((long)(3 * BB + b) * LQ + (LQ - 1 - n1)) * DMQ;

    float v1 = dirout[r0 + d] + dirout[r1 + d];
    float v2 = dirout[r2 + d] + dirout[r3 + d];

    __shared__ float red[256];
    auto blockSum = [&](float v) -> float {
        red[threadIdx.x] = v; __syncthreads();
        for (int s = 128; s > 0; s >>= 1) {
            if (threadIdx.x < s) red[threadIdx.x] += red[threadIdx.x + s];
            __syncthreads();
        }
        float r = red[0]; __syncthreads();
        return r;
    };

    float m1 = blockSum(v1) * (1.f / DMQ);
    float q1 = blockSum(v1 * v1) * (1.f / DMQ);
    float m2 = blockSum(v2) * (1.f / DMQ);
    float q2 = blockSum(v2 * v2) * (1.f / DMQ);
    float inv1 = rsqrtf(q1 - m1 * m1 + 1e-5f);
    float inv2 = rsqrtf(q2 - m2 * m2 + 1e-5f);
    float s1 = (v1 - m1) * inv1 * lnw[d] + lnb[d];
    float s2 = (v2 - m2) * inv2 * lnw[d] + lnb[d];

    long xi = (((long)b * OO + o) * TT + t) * DMQ + d;
    x[xi] = 0.5f * (s1 + s2) + x[xi];
}

// =====================================================================
extern "C" void kernel_launch(void* const* d_in, const int* in_sizes, int n_in,
                              void* d_out, int out_size, void* d_ws, size_t ws_size,
                              hipStream_t stream) {
    (void)in_sizes; (void)n_in; (void)ws_size;

    const float* x_in  = (const float*)d_in[0];
    const float* pe    = (const float*)d_in[1];
    const float* ln_w  = (const float*)d_in[2];
    const float* ln_b  = (const float*)d_in[3];
    const float* inp   = (const float*)d_in[4];
    const float* convw = (const float*)d_in[5];
    const float* convb = (const float*)d_in[6];
    const float* xproj = (const float*)d_in[7];
    const float* dtw   = (const float*)d_in[8];
    const float* dtb   = (const float*)d_in[9];
    const float* alog  = (const float*)d_in[10];
    const float* Dp    = (const float*)d_in[11];
    const float* outp  = (const float*)d_in[12];

    float* x = (float*)d_out;                 // running activation (B,O,T,DM)

    // workspace carve-up (f32)
    float* ws = (float*)d_ws;
    long off = 0;
    float* seq    = ws + off; off += (long)NCOMBO * MPC * DMQ;            // 6.29M
    float* xz     = ws + off; off += (long)NCOMBO * MPC * 2 * DIQ;        // 25.2M
    float* xc     = ws + off; off += (long)NCOMBO * MPC * DIQ;            // 12.6M
    float* dbl    = ws + off; off += (long)NCOMBO * MPC * (DTRQ + 2*DSQ); // 1.18M
    float* delta  = ws + off; off += (long)NCOMBO * MPC * DIQ;            // 12.6M
    float* ybuf   = ws + off; off += (long)NCOMBO * MPC * DIQ;            // 12.6M
    float* dirout = ws + off; off += (long)NCOMBO * MPC * DMQ;            // 6.29M

    // x = input + positional embedding
    {
        int n = out_size;
        add_pe_kernel<<<(n + 255) / 256, 256, 0, stream>>>(x_in, pe, x, n);
    }

    for (int l = 0; l < NLQ; ++l) {
        const float* inp_l  = inp   + (long)l * NCOMBO * DMQ * 2 * DIQ;
        const float* cw_l   = convw + (long)l * NCOMBO * DIQ * DCONVQ;
        const float* cb_l   = convb + (long)l * NCOMBO * DIQ;
        const float* xp_l   = xproj + (long)l * NCOMBO * DIQ * (DTRQ + 2*DSQ);
        const float* dtw_l  = dtw   + (long)l * NCOMBO * DTRQ * DIQ;
        const float* dtb_l  = dtb   + (long)l * NCOMBO * DIQ;
        const float* al_l   = alog  + (long)l * NCOMBO * DIQ * DSQ;
        const float* dp_l   = Dp    + (long)l * NCOMBO * DIQ;
        const float* op_l   = outp  + (long)l * NCOMBO * DIQ * DMQ;

        // gather 4 combo sequences from x
        {
            long n = (long)NCOMBO * MPC * DMQ;
            gather_seq_kernel<<<(int)((n + 255) / 256), 256, 0, stream>>>(x, seq);
        }
        // in_proj GEMM: (6144x256)@(256x1024) per combo
        gemm_wmma_kernel<<<dim3((2*DIQ)/64, MPC/64, NCOMBO), 128, 0, stream>>>(
            seq, (long)MPC * DMQ, DMQ,
            inp_l, (long)DMQ * 2 * DIQ,
            nullptr, 0,
            xz, (long)MPC * 2 * DIQ, 2 * DIQ,
            2 * DIQ, DMQ, 0);
        // conv + silu
        {
            long n = (long)NCOMBO * MPC * DIQ;
            conv_silu_kernel<<<(int)((n + 255) / 256), 256, 0, stream>>>(xz, cw_l, cb_l, xc);
        }
        // x_proj GEMM: (6144x512)@(512x48)
        gemm_wmma_kernel<<<dim3(1, MPC/64, NCOMBO), 128, 0, stream>>>(
            xc, (long)MPC * DIQ, DIQ,
            xp_l, (long)DIQ * (DTRQ + 2*DSQ),
            nullptr, 0,
            dbl, (long)MPC * (DTRQ + 2*DSQ), DTRQ + 2*DSQ,
            DTRQ + 2*DSQ, DIQ, 0);
        // dt GEMM: (6144x16)@(16x512) + bias, softplus -> delta
        gemm_wmma_kernel<<<dim3(DIQ/64, MPC/64, NCOMBO), 128, 0, stream>>>(
            dbl, (long)MPC * (DTRQ + 2*DSQ), DTRQ + 2*DSQ,
            dtw_l, (long)DTRQ * DIQ,
            dtb_l, (long)DIQ,
            delta, (long)MPC * DIQ, DIQ,
            DIQ, DTRQ, 1);
        // selective scan
        scan_kernel<<<NCOMBO * BB * 2, 256, 0, stream>>>(delta, xc, dbl, al_l, dp_l, ybuf);
        // y *= silu(z)
        {
            long n = (long)NCOMBO * MPC * DIQ;
            gate_kernel<<<(int)((n + 255) / 256), 256, 0, stream>>>(xz, ybuf);
        }
        // out_proj GEMM: (6144x512)@(512x256)
        gemm_wmma_kernel<<<dim3(DMQ/64, MPC/64, NCOMBO), 128, 0, stream>>>(
            ybuf, (long)MPC * DIQ, DIQ,
            op_l, (long)DIQ * DMQ,
            nullptr, 0,
            dirout, (long)MPC * DMQ, DMQ,
            DMQ, DIQ, 0);
        // combine dirs/blks, layernorm, average, residual (in-place on x)
        combine_ln_kernel<<<BB * OO * TT, 256, 0, stream>>>(dirout, ln_w, ln_b, x);
    }
}
